// GQA_37125697306882
// MI455X (gfx1250) — matrix operand order repaired and checked by reading, not compiled, not run on hardware
//
#include <hip/hip_runtime.h>

// ---------------------------------------------------------------------------
// GQA layer for MI455X (gfx1250).
// Pipeline:
//   1) convert x, Wq,Wk,Wv,Wo  f32 -> bf16 in workspace (W's concatenated)
//   2) QKV GEMM  : [16384,2048] x [6144,2048]^T -> qkv bf16   (WMMA bf16,
//      3-stage async pipeline: global_load_async_to_lds_b128, 3 LDS buffers,
//      two tile-groups kept in flight via s_wait_asynccnt 4)
//   3) RoPE + head-permute + per-token 16x16 attention (f32 LDS) -> attn bf16
//   4) OUT GEMM  : [16384,2048] x [2048,2048]^T -> d_out f32   (WMMA bf16)
// ---------------------------------------------------------------------------

typedef unsigned short u16;
typedef __attribute__((ext_vector_type(16))) __bf16 v16bf;
typedef __attribute__((ext_vector_type(8)))  __bf16 v8bf;
typedef __attribute__((ext_vector_type(8)))  float  v8f;
typedef __attribute__((ext_vector_type(4)))  float  f32x4;
typedef __attribute__((ext_vector_type(8)))  u16    u16x8;

union Frag16 { v16bf v; v8bf h[2]; };

__device__ __forceinline__ u16 f2bf(float f) {
  unsigned u = __float_as_uint(f);
  unsigned r = u + 0x7FFFu + ((u >> 16) & 1u);   // round-to-nearest-even
  return (u16)(r >> 16);
}
__device__ __forceinline__ float bf2f(u16 b) {
  return __uint_as_float(((unsigned)b) << 16);
}

// CDNA5 async DMA: memory -> LDS, 16B per lane, tracked by ASYNCcnt.
// LDS generic pointer truncated to 32 bits == wave-relative LDS byte offset.
__device__ __forceinline__ void async_b128(const u16* gsrc, u16* lds_dst) {
  const unsigned loff = (unsigned)(size_t)lds_dst;
  const unsigned long long g = (unsigned long long)(size_t)gsrc;
  asm volatile("global_load_async_to_lds_b128 %0, %1, off"
               :: "v"(loff), "v"(g) : "memory");
}
// async loads retire in order: waiting to <=4 retires all but the youngest
// 4-op tile group; waiting to 0 drains everything (tail iterations).
__device__ __forceinline__ void wait_async4() {
  asm volatile("s_wait_asynccnt 0x4" ::: "memory");
}
__device__ __forceinline__ void wait_async0() {
  asm volatile("s_wait_asynccnt 0x0" ::: "memory");
}

// ---------------------------------------------------------------------------
// 1) f32 -> bf16 converter (8 elems / thread, vectorized)
// ---------------------------------------------------------------------------
__global__ void __launch_bounds__(256)
f32_to_bf16_kernel(const float* __restrict__ src, u16* __restrict__ dst, int n) {
  int i = (blockIdx.x * 256 + threadIdx.x) * 8;
  if (i + 8 > n) return;
  f32x4 a = *(const f32x4*)(src + i);
  f32x4 b = *(const f32x4*)(src + i + 4);
  u16x8 o;
  o[0] = f2bf(a[0]); o[1] = f2bf(a[1]); o[2] = f2bf(a[2]); o[3] = f2bf(a[3]);
  o[4] = f2bf(b[0]); o[5] = f2bf(b[1]); o[6] = f2bf(b[2]); o[7] = f2bf(b[3]);
  *(u16x8*)(dst + i) = o;
}

// ---------------------------------------------------------------------------
// 2/4) Tiled bf16 WMMA GEMM:  C[M,N] = A[M,K] * B[N,K]^T
//   block tile 128x128, BK=32, 256 threads = 8 waves in a 2(M) x 4(N) grid,
//   each wave computes 64x32 (4x2 fragments of 16x16, f32 accum).
//   3 LDS buffers, rows padded to 40 halves (80B): 16B-aligned chunks and a
//   conflict-free bank rotation for fragment reads.  Staging two K-tiles
//   ahead via global_load_async_to_lds_b128 (ASYNCcnt).  Buffer rotation is
//   hand-unrolled (x3) so every LDS offset is a compile-time immediate and
//   the steady-state loop has no branches.
// ---------------------------------------------------------------------------
#define BM 128
#define BN 128
#define BK 32
#define LSTR 40   // halves per LDS row

template <int K, bool C_BF16>
__global__ void __launch_bounds__(256)
gemm_bf16_wmma_kernel(const u16* __restrict__ A, const u16* __restrict__ B,
                      void* __restrict__ C, int ldc) {
  constexpr int NK = K / BK;
  static_assert(NK >= 5 && (NK - 2) % 3 == 2, "tail peeling assumes this NK");

  __shared__ u16 At[3][BM * LSTR];
  __shared__ u16 Bt[3][BN * LSTR];

  const int tid  = threadIdx.x;
  const int lane = tid & 31;
  const int wave = tid >> 5;
  const int wm   = wave >> 2;      // 0..1 : 64-row strip
  const int wn   = wave & 3;       // 0..3 : 32-col strip
  const int hh   = lane >> 4;      // lane half (K-interleave select)
  const int lr   = lane & 15;      // row/col within fragment
  const size_t m0 = (size_t)blockIdx.y * BM;
  const size_t n0 = (size_t)blockIdx.x * BN;

  // loader mapping: thread t owns 32 contiguous bytes of tile row (t>>1)
  const int lrow = tid >> 1;
  const int loff = (tid & 1) * 16;           // halves
  const u16* gA = A + (m0 + lrow) * K + loff;
  const u16* gB = B + (n0 + lrow) * K + loff;
  u16* lA[3] = { &At[0][lrow * LSTR + loff], &At[1][lrow * LSTR + loff],
                 &At[2][lrow * LSTR + loff] };
  u16* lB[3] = { &Bt[0][lrow * LSTR + loff], &Bt[1][lrow * LSTR + loff],
                 &Bt[2][lrow * LSTR + loff] };

  v8f acc[4][2] = {};

  // stage K-tile 'ktile' into LDS buffer 'sbuf' (4 async ops / thread)
  auto stage = [&](int sbuf, int ktile) {
    const int k2 = ktile * BK;
    async_b128(gA + k2, lA[sbuf]);  async_b128(gA + k2 + 8, lA[sbuf] + 8);
    async_b128(gB + k2, lB[sbuf]);  async_b128(gB + k2 + 8, lB[sbuf] + 8);
  };

  // consume LDS buffer 'cbuf': 12 ds_load_b128 + 8 WMMAs
  auto compute = [&](int cbuf) {
    // B frags: lane lr = column n, contiguous K chunk [hh*16, +16)
    Frag16 bfr[2];
    #pragma unroll
    for (int j = 0; j < 2; ++j) {
      const u16* p = &Bt[cbuf][(wn * 32 + j * 16 + lr) * LSTR + hh * 16];
      bfr[j].h[0] = *(const v8bf*)(p);
      bfr[j].h[1] = *(const v8bf*)(p + 8);
    }
    // A frags: lane lr = row m, contiguous K chunk [hh*16, +16)
    Frag16 afr[4];
    #pragma unroll
    for (int i = 0; i < 4; ++i) {
      const u16* p = &At[cbuf][(wm * 64 + i * 16 + lr) * LSTR];
      afr[i].h[0] = *(const v8bf*)(p + hh * 16);
      afr[i].h[1] = *(const v8bf*)(p + hh * 16 + 8);
    }
    #pragma unroll
    for (int i = 0; i < 4; ++i) {
      #pragma unroll
      for (int j = 0; j < 2; ++j) {
        acc[i][j] = __builtin_amdgcn_wmma_f32_16x16x32_bf16(
            false, afr[i].v, false, bfr[j].v, (short)0, acc[i][j], false, false);
      }
    }
  };

  // prologue: tiles 0 and 1 in flight; wait until tile 0 resident
  stage(0, 0);
  stage(1, 1);
  wait_async4();
  __syncthreads();

  // steady state: buffer indices are literals in each of the 3 bodies
  int kt = 0;
  #pragma unroll 1
  for (; kt + 3 <= NK - 2; kt += 3) {
    stage(2, kt + 2); compute(0); wait_async4(); __syncthreads();
    stage(0, kt + 3); compute(1); wait_async4(); __syncthreads();
    stage(1, kt + 4); compute(2); wait_async4(); __syncthreads();
  }
  // (NK-2) % 3 == 2 remainder staged steps  (kt == NK-4 here)
  stage(2, kt + 2); compute(0); wait_async4(); __syncthreads();
  stage(0, kt + 3); compute(1); wait_async0(); __syncthreads();
  // tails: tiles NK-2 (buffer 2) and NK-1 (buffer 0), nothing left in flight
  compute(2);
  __syncthreads();
  compute(0);

  // ---- writeback: C/D layout — lane lr = col n, VGPR r = row (r + 8*hh)
  #pragma unroll
  for (int i = 0; i < 4; ++i) {
    #pragma unroll
    for (int j = 0; j < 2; ++j) {
      const size_t mb = m0 + wm * 64 + i * 16 + hh * 8;
      const size_t nn = n0 + wn * 32 + j * 16 + lr;
      if constexpr (C_BF16) {
        u16* Cp = (u16*)C;
        #pragma unroll
        for (int r = 0; r < 8; ++r) Cp[(mb + r) * ldc + nn] = f2bf(acc[i][j][r]);
      } else {
        float* Cp = (float*)C;
        #pragma unroll
        for (int r = 0; r < 8; ++r) Cp[(mb + r) * ldc + nn] = acc[i][j][r];
      }
    }
  }
}

// ---------------------------------------------------------------------------
// 3) Per-token RoPE + head permute + 16x16 attention (f32 in LDS).
//    qkv layout: [tok, 6144] bf16 = q(2048) | k(2048) | v(2048)
// ---------------------------------------------------------------------------
__global__ void __launch_bounds__(128)
rope_attn_kernel(const u16* __restrict__ qkv, u16* __restrict__ attn_out) {
  __shared__ float qs[16 * 128];
  __shared__ float ks[16 * 128];
  __shared__ float vs[16 * 132];   // pad to 132 to break 128-stride conflicts
  __shared__ float sc[16 * 17];
  __shared__ float pr[16 * 17];

  const int tok = blockIdx.x;
  const int s   = tok & 4095;              // position within sequence (S=4096)
  const int tid = threadIdx.x;
  const u16* base = qkv + (size_t)tok * 6144;

  // ---- load + RoPE (q with 4x4 head-transpose permute, k plain), v copy ----
  #pragma unroll 4
  for (int it = 0; it < 16; ++it) {
    const int e = tid + it * 128;          // 0..2047
    const int h = e >> 7;
    const int d = e & 127;
    const int i = d >> 1;
    // inv_freq = 10000^(-i/64) ; log2(10000)/64 = 0.2076205059...
    const float freq = exp2f((float)i * -0.20762050593f);
    const float ang  = (float)s * freq;
    float cs, sn;
    __sincosf(ang, &sn, &cs);
    {
      // permutation applied AFTER RoPE in the reference; RoPE is head-
      // independent so they commute: new head h  <-  old head (h%4)*4 + h/4
      const int oh = (h & 3) * 4 + (h >> 2);
      const int qb = oh * 128 + (d & ~1);
      const float q1 = bf2f(base[qb]), q2 = bf2f(base[qb + 1]);
      qs[e] = ((d & 1) == 0) ? (q1 * cs - q2 * sn) : (q1 * sn + q2 * cs);
    }
    {
      const int kb = 2048 + h * 128 + (d & ~1);
      const float k1 = bf2f(base[kb]), k2 = bf2f(base[kb + 1]);
      ks[e] = ((d & 1) == 0) ? (k1 * cs - k2 * sn) : (k1 * sn + k2 * cs);
    }
    vs[h * 132 + d] = bf2f(base[4096 + e]);
  }
  __syncthreads();

  // ---- scores[h,t] = (q_h . k_t) / sqrt(128) ----
  #pragma unroll
  for (int j = 0; j < 2; ++j) {
    const int e = tid + j * 128;           // 0..255
    const int h = e >> 4, t = e & 15;
    float a = 0.f;
    #pragma unroll 8
    for (int kk = 0; kk < 128; ++kk) a += qs[h * 128 + kk] * ks[t * 128 + kk];
    sc[h * 17 + t] = a * 0.08838834764831845f;
  }
  __syncthreads();

  // ---- softmax over t (f32, one thread per head row) ----
  if (tid < 16) {
    const int h = tid;
    float mx = -3.4e38f;
    for (int t = 0; t < 16; ++t) mx = fmaxf(mx, sc[h * 17 + t]);
    float sum = 0.f;
    for (int t = 0; t < 16; ++t) {
      const float p = __expf(sc[h * 17 + t] - mx);
      pr[h * 17 + t] = p;
      sum += p;
    }
    const float inv = 1.f / sum;
    for (int t = 0; t < 16; ++t) pr[h * 17 + t] *= inv;
  }
  __syncthreads();

  // ---- attn[h,d] = sum_t p[h,t] * v[t,d]  -> bf16 ----
  #pragma unroll 4
  for (int it = 0; it < 16; ++it) {
    const int e = tid + it * 128;
    const int h = e >> 7, d = e & 127;
    float a = 0.f;
    #pragma unroll
    for (int t = 0; t < 16; ++t) a += pr[h * 17 + t] * vs[t * 132 + d];
    attn_out[(size_t)tok * 2048 + e] = f2bf(a);
  }
}

// ---------------------------------------------------------------------------
// launcher
// ---------------------------------------------------------------------------
extern "C" void kernel_launch(void* const* d_in, const int* in_sizes, int n_in,
                              void* d_out, int out_size, void* d_ws, size_t ws_size,
                              hipStream_t stream) {
  const float* x  = (const float*)d_in[0];
  const float* Wq = (const float*)d_in[1];
  const float* Wk = (const float*)d_in[2];
  const float* Wv = (const float*)d_in[3];
  const float* Wo = (const float*)d_in[4];

  char* ws = (char*)d_ws;
  const size_t X_ELE = 16384ull * 2048ull;   // 33,554,432
  const size_t W_ELE = 2048ull * 2048ull;    //  4,194,304
  u16* xb   = (u16*)(ws);                                  // 64 MB
  u16* wb   = (u16*)(ws + 2 * X_ELE);                      // Wq|Wk|Wv|Wo, 32 MB
  u16* qkv  = (u16*)(ws + 2 * X_ELE + 2 * 4 * W_ELE);      // [16384,6144] bf16
  u16* attb = (u16*)(ws + 2 * X_ELE + 2 * 4 * W_ELE + 2ull * 16384 * 6144);

  // 1) conversions
  f32_to_bf16_kernel<<<(int)(X_ELE / 2048), 256, 0, stream>>>(x, xb, (int)X_ELE);
  f32_to_bf16_kernel<<<(int)(W_ELE / 2048), 256, 0, stream>>>(Wq, wb + 0 * W_ELE, (int)W_ELE);
  f32_to_bf16_kernel<<<(int)(W_ELE / 2048), 256, 0, stream>>>(Wk, wb + 1 * W_ELE, (int)W_ELE);
  f32_to_bf16_kernel<<<(int)(W_ELE / 2048), 256, 0, stream>>>(Wv, wb + 2 * W_ELE, (int)W_ELE);
  f32_to_bf16_kernel<<<(int)(W_ELE / 2048), 256, 0, stream>>>(Wo, wb + 3 * W_ELE, (int)W_ELE);

  // 2) fused QKV projection: N = 6144 (Wq|Wk|Wv rows contiguous) -> bf16
  dim3 gq(6144 / BN, 16384 / BM);
  gemm_bf16_wmma_kernel<2048, true><<<gq, 256, 0, stream>>>(xb, wb, (void*)qkv, 6144);

  // 3) RoPE + head permute + per-token attention -> bf16
  rope_attn_kernel<<<16384, 128, 0, stream>>>(qkv, attb);

  // 4) output projection -> f32 d_out
  dim3 go(2048 / BN, 16384 / BM);
  gemm_bf16_wmma_kernel<2048, false><<<go, 256, 0, stream>>>(attb, wb + 3 * W_ELE, d_out, 2048);
}
